// Tree_42520176230890
// MI455X (gfx1250) — compile-verified, matrix operand order checked by reading, not run on hardware
//
#include <hip/hip_runtime.h>

typedef __attribute__((ext_vector_type(16))) __bf16 v16bf;
typedef __attribute__((ext_vector_type(8)))  float  v8f;

#define FEAT 1024
#define NLEAF 1024
#define BATCH 16384
#define OUTW 2048

// ---------------------------------------------------------------------------
// Kernel 1: recover using_idx[j] from one-hot feature_mask (FEAT x NLEAF).
// ---------------------------------------------------------------------------
__global__ __launch_bounds__(256) void build_idx_kernel(
    const float* __restrict__ mask, int* __restrict__ idx) {
  int j = blockIdx.x * 256 + threadIdx.x;
  if (j >= NLEAF) return;
  int found = 0;
  for (int i = 0; i < FEAT; ++i) {
    if (mask[(size_t)i * NLEAF + j] != 0.0f) found = i;
  }
  idx[j] = found;
}

// Fragment slot mapping (16-bit WMMA operand layout, ISA 7.12.2):
// lane half h = lane>>4 ; element j (0..15) ->
//   kk = (h<<3) | (j&7) | ((j&8)<<1)     (K index within 32-wide tile)
// ---------------------------------------------------------------------------
// Kernel 2: gather x columns via idx, cast to bf16, store A-fragment-major.
// Afrag layout: [m_tile(1024)][k_tile(32)][lane(32)][j(16)]
// ---------------------------------------------------------------------------
__global__ __launch_bounds__(256) void gather_cast_A_kernel(
    const float* __restrict__ x, const int* __restrict__ idx,
    __bf16* __restrict__ Afrag) {
  unsigned e = blockIdx.x * 256u + threadIdx.x;   // 16384*1024 elements
  int j      = e & 15;
  int lane   = (e >> 4) & 31;
  int k_tile = (e >> 9) & 31;
  int m_tile = e >> 14;
  int h  = lane >> 4;
  int kk = (h << 3) | (j & 7) | ((j & 8) << 1);
  int col = k_tile * 32 + kk;                     // leaf/feature slot
  int row = m_tile * 16 + (lane & 15);            // batch row (M)
  float v = x[(size_t)row * FEAT + idx[col]];
  Afrag[e] = (__bf16)v;
}

// ---------------------------------------------------------------------------
// Kernel 3: cast W (K x N, row-major) to bf16, store B-fragment-major.
// Bfrag layout: [k_tile(32)][n_tile(64)][lane(32)][j(16)]
// ---------------------------------------------------------------------------
__global__ __launch_bounds__(256) void cast_B_kernel(
    const float* __restrict__ W, __bf16* __restrict__ Bfrag) {
  unsigned e = blockIdx.x * 256u + threadIdx.x;   // 1024*1024 elements
  int j      = e & 15;
  int lane   = (e >> 4) & 31;
  int n_tile = (e >> 9) & 63;
  int k_tile = e >> 15;
  int h  = lane >> 4;
  int kk = (h << 3) | (j & 7) | ((j & 8) << 1);
  int k = k_tile * 32 + kk;
  int n = n_tile * 16 + (lane & 15);
  Bfrag[e] = (__bf16)W[(size_t)k * NLEAF + n];
}

// ---------------------------------------------------------------------------
// Kernel 4: fused GEMM (bf16 WMMA) + sigmoid + soft tree epilogue.
// Block = 32 batch rows x 1024 leaves, 512 threads = 16 waves.
// Each wave: 2 M-tiles x 4 N-tiles (64 acc VGPRs); every B fragment feeds
// two WMMAs. Explicit ping-pong double buffering (X/Y fragment sets) keeps
// ~12 b128 loads in flight behind each WMMA burst -> partial loadcnt waits
// instead of wait-0 stalls. ~190 VGPRs total: no MSB banks, no mov storms.
// d block (32x1024) staged in 128KB LDS; tree products read d from LDS.
// ---------------------------------------------------------------------------
__global__ __launch_bounds__(512) void gemm_tree_kernel(
    const __bf16* __restrict__ Afrag,
    const __bf16* __restrict__ Bfrag,
    const float* __restrict__ bias,
    float* __restrict__ out) {
  __shared__ float d_s[32 * NLEAF];               // 128 KB

  const int tid  = threadIdx.x;
  const int wave = tid >> 5;                      // 0..15
  const int lane = tid & 31;
  const int m_blk = blockIdx.x;                   // 32 rows per block

  v8f acc0[4], acc1[4];
#pragma unroll
  for (int t = 0; t < 4; ++t) {
    acc0[t] = (v8f){0.f, 0.f, 0.f, 0.f, 0.f, 0.f, 0.f, 0.f};
    acc1[t] = (v8f){0.f, 0.f, 0.f, 0.f, 0.f, 0.f, 0.f, 0.f};
  }

  const __bf16* a0Base = Afrag + ((size_t)(m_blk * 2 + 0) * 32) * 512 + lane * 16;
  const __bf16* a1Base = Afrag + ((size_t)(m_blk * 2 + 1) * 32) * 512 + lane * 16;
  const __bf16* bBase  = Bfrag + (size_t)(wave * 4) * 512 + lane * 16;
  // byte pointer to this wave's 4KB B slab for lane-strided prefetch
  const char* bPf = (const char*)(Bfrag + (size_t)(wave * 4) * 512) + (size_t)lane * 128;

  // preload k-step 0 into buffer X
  v16bf a0X = *reinterpret_cast<const v16bf*>(a0Base);
  v16bf a1X = *reinterpret_cast<const v16bf*>(a1Base);
  v16bf bX[4];
#pragma unroll
  for (int t = 0; t < 4; ++t)
    bX[t] = *reinterpret_cast<const v16bf*>(bBase + t * 512);

  for (int kt = 0; kt < 32; kt += 2) {
    // prefetch 3 k-steps ahead (clamped)
    int kpf = (kt + 3 <= 31) ? (kt + 3) : 31;
    __builtin_prefetch(bPf + (size_t)kpf * (64 * 512 * 2), 0, 3);

    // issue buffer Y loads (k-step kt+1)
    const int kY = kt + 1;
    v16bf a0Y = *reinterpret_cast<const v16bf*>(a0Base + kY * 512);
    v16bf a1Y = *reinterpret_cast<const v16bf*>(a1Base + kY * 512);
    v16bf bY[4];
#pragma unroll
    for (int t = 0; t < 4; ++t)
      bY[t] = *reinterpret_cast<const v16bf*>(bBase + (size_t)kY * (64 * 512) + t * 512);

    // WMMA burst on X (loads were issued one half-step ago)
#pragma unroll
    for (int t = 0; t < 4; ++t) {
      acc0[t] = __builtin_amdgcn_wmma_f32_16x16x32_bf16(
          false, a0X, false, bX[t], (short)0, acc0[t], false, false);
      acc1[t] = __builtin_amdgcn_wmma_f32_16x16x32_bf16(
          false, a1X, false, bX[t], (short)0, acc1[t], false, false);
    }

    // issue buffer X loads (k-step kt+2, clamped; dead on last iteration)
    const int kX = (kt + 2 <= 31) ? (kt + 2) : 31;
    a0X = *reinterpret_cast<const v16bf*>(a0Base + kX * 512);
    a1X = *reinterpret_cast<const v16bf*>(a1Base + kX * 512);
#pragma unroll
    for (int t = 0; t < 4; ++t)
      bX[t] = *reinterpret_cast<const v16bf*>(bBase + (size_t)kX * (64 * 512) + t * 512);

    // WMMA burst on Y
#pragma unroll
    for (int t = 0; t < 4; ++t) {
      acc0[t] = __builtin_amdgcn_wmma_f32_16x16x32_bf16(
          false, a0Y, false, bY[t], (short)0, acc0[t], false, false);
      acc1[t] = __builtin_amdgcn_wmma_f32_16x16x32_bf16(
          false, a1Y, false, bY[t], (short)0, acc1[t], false, false);
    }
  }

  // Epilogue: d = sigmoid(acc + bias) -> LDS.
  // C layout: lanes 0-15 hold N=lane, rows M=r (VGPR r); lanes 16-31 M=8+r.
  const int ncol = lane & 15;
  const int mrow_base = (lane >> 4) * 8;
#pragma unroll
  for (int t = 0; t < 4; ++t) {
    int n = wave * 64 + t * 16 + ncol;
    float bv = bias[n];
#pragma unroll
    for (int r = 0; r < 8; ++r) {
      float v0 = acc0[t][r] + bv;
      float v1 = acc1[t][r] + bv;
      d_s[(mrow_base + r) * NLEAF + n]      = 1.0f / (1.0f + __expf(-v0));
      d_s[(16 + mrow_base + r) * NLEAF + n] = 1.0f / (1.0f + __expf(-v1));
    }
  }
  __syncthreads();

  // Soft decision tree: out[row, e].
  // Layout: e in [0,2) -> 1.0 ; layer l occupies [2^(l+1), 2^(l+2)),
  // value = prod_{j=0..l} dec(node_j, s_j),
  //   node_j = 2^j + (k >> (l-j+1)), s_j = (k >> (l-j)) & 1,
  //   dec(n,0)=d[n], dec(n,1)=1-d[n].
  const size_t outBase = (size_t)m_blk * 32 * OUTW;
  for (int it = tid; it < 32 * OUTW; it += 512) {
    int r = it >> 11;
    int e = it & (OUTW - 1);
    float val;
    if (e < 2) {
      val = 1.0f;
    } else {
      int l = 30 - __clz(e);           // layer index 0..9
      int k = e - (1 << (l + 1));      // index within layer
      float prod = 1.0f;
      for (int j = 0; j <= l; ++j) {
        int shift = l - j;
        int node = (1 << j) + (k >> (shift + 1));
        int s = (k >> shift) & 1;
        float dv = d_s[r * NLEAF + node];
        prod *= s ? (1.0f - dv) : dv;
      }
      val = prod;
    }
    out[outBase + (size_t)r * OUTW + e] = val;
  }
}

// ---------------------------------------------------------------------------
extern "C" void kernel_launch(void* const* d_in, const int* in_sizes, int n_in,
                              void* d_out, int out_size, void* d_ws, size_t ws_size,
                              hipStream_t stream) {
  const float* x    = (const float*)d_in[0];   // (16384, 1024)
  const float* mask = (const float*)d_in[1];   // (1024, 1024)
  const float* W    = (const float*)d_in[2];   // (1024, 1024)
  const float* bias = (const float*)d_in[3];   // (1024,)
  float* out = (float*)d_out;                  // (16384, 2048)

  // Workspace layout
  char* ws = (char*)d_ws;
  int*    idx   = (int*)ws;                          //    4 KB
  __bf16* Afrag = (__bf16*)(ws + 4096);              //   32 MB (1024*32*512 bf16)
  __bf16* Bfrag = (__bf16*)(ws + 4096 + (size_t)BATCH * FEAT * 2); // 2 MB

  build_idx_kernel<<<4, 256, 0, stream>>>(mask, idx);
  gather_cast_A_kernel<<<(BATCH * FEAT) / 256, 256, 0, stream>>>(x, idx, Afrag);
  cast_B_kernel<<<(NLEAF * NLEAF) / 256, 256, 0, stream>>>(W, Bfrag);
  gemm_tree_kernel<<<BATCH / 32, 512, 0, stream>>>(Afrag, Bfrag, bias, out);
}
